// GNNLayer6_39410619908404
// MI455X (gfx1250) — compile-verified, matrix operand order checked by reading, not run on hardware
//
#include <hip/hip_runtime.h>
#include <hip/hip_bf16.h>

typedef __attribute__((ext_vector_type(16))) __bf16 v16bf;
typedef __attribute__((ext_vector_type(8)))  float  v8f;

#define IN_DIM   128
#define ATTN_DIM 64
#define N_NODE   50000
#define N_EDGE   500000
#define EDGE_TILES (N_EDGE / 16)   // 31250
#define NODE_TILES (N_NODE / 16)   // 3125

// ---- workspace layout (bytes) ----
#define OFF_AGG   0ull                         // 50000*128*4 = 25,600,000
#define OFF_DEG   25600000ull                  // 50000*4     =    200,000
#define OFF_GATEW 25800000ull                  // 256*384*2   =    196,608
#define OFF_HT    25996608ull                  // 128*256*2   =     65,536
#define OFF_WS    26062144ull                  // 64*128*2    =     16,384
#define OFF_WR    26078528ull
#define OFF_WQR   26094912ull
#define OFF_WH    26111296ull                  // 128*128*2   =     32,768
// total = 26,144,064 bytes

#define WAVE_LDS 57472   // per-wave LDS slice in edge kernel (16B aligned)

__device__ __forceinline__ unsigned short f2bf(float f) {
    unsigned u = __float_as_uint(f);
    return (unsigned short)((u + 0x7FFFu + ((u >> 16) & 1u)) >> 16);
}
__device__ __forceinline__ unsigned pack2(float a, float b) {
    return (unsigned)f2bf(a) | ((unsigned)f2bf(b) << 16);
}

union FragCast { uint4 q[2]; v16bf v; };

// A/B fragment loader for V_WMMA_*_16x16x32 bf16 (row-major source, stride ld elems).
// Lane l (l<16): row=l, K = kbase+[0..7] and kbase+16+[0..7];
// Lane l (l>=16): row=l-16, K = kbase+8+[0..7] and kbase+24+[0..7].
__device__ __forceinline__ v16bf load_frag(const unsigned short* base, int ld, int kbase, int lane) {
    const unsigned short* p = base + (lane & 15) * ld + kbase + (lane >> 4) * 8;
    FragCast f;
    f.q[0] = *(const uint4*)(p);
    f.q[1] = *(const uint4*)(p + 16);
    return f.v;
}

#define WMMA_BF16(A, B, C) \
    __builtin_amdgcn_wmma_f32_16x16x32_bf16(false, (A), false, (B), (short)0, (C), false, false)

__device__ __forceinline__ float sigm(float x) { return 1.0f / (1.0f + __expf(-x)); }

// ---------------- prep kernels ----------------
__global__ void zero_f32_kernel(float* p, int n) {
    for (int i = blockIdx.x * blockDim.x + threadIdx.x; i < n; i += gridDim.x * blockDim.x)
        p[i] = 0.0f;
}

__global__ void cvt_bf16_kernel(const float* __restrict__ src, unsigned short* __restrict__ dst, int n) {
    int i = blockIdx.x * blockDim.x + threadIdx.x;
    if (i < n) dst[i] = f2bf(src[i]);
}

// ---------------- edge kernel: per-edge GRU-ish message + attention + scatter ----------------
__global__ void __launch_bounds__(128) edge_kernel(
    const int* __restrict__ q_rel, const float* __restrict__ hidden,
    const long long* __restrict__ edges, const float* __restrict__ rela,
    const float* __restrict__ Wqr_b, const float* __restrict__ walpha,
    const float* __restrict__ gateW_b, const float* __restrict__ ht_b,
    const unsigned short* __restrict__ gateWbf, const unsigned short* __restrict__ htbf,
    const unsigned short* __restrict__ Wsbf, const unsigned short* __restrict__ Wrbf,
    const unsigned short* __restrict__ Wqrbf,
    float* __restrict__ agg, float* __restrict__ deg)
{
    extern __shared__ char smem[];
    const int lane = threadIdx.x & 31;
    const int wave = threadIdx.x >> 5;
    const int tile = blockIdx.x * 4 + wave;
    const bool active = tile < EDGE_TILES;

    char* wb = smem + (size_t)wave * WAVE_LDS;
    unsigned short* factors = (unsigned short*)(wb);          // 16 x 384 bf16 = 12288 B  [hr|h_qr|hs]
    unsigned short* candin  = (unsigned short*)(wb + 12288);  // 16 x 256 bf16 =  8192 B  [hr|reset*hs]
    float* hsf    = (float*)(wb + 20480);                     // 16 x 128 f32  =  8192 B
    float* gatesL = (float*)(wb + 28672);                     // 16 x 256 f32  = 16384 B
    float* msgL   = (float*)(wb + 45056);                     // 16 x 128 f32  =  8192 B
    float* preL   = (float*)(wb + 53248);                     // 16 x  64 f32  =  4096 B
    float* alphaL = (float*)(wb + 57344);                     // 16 f32

    // ---- load edge indices (one edge per lane 0..15) ----
    int subv = 0, relv = 0, qrv = 0, objv = 0;
    if (active && lane < 16) {
        long long eb = ((long long)tile * 16 + lane) * 7;
        qrv  = (int)edges[eb + 0];
        relv = (int)edges[eb + 2];
        subv = (int)edges[eb + 4];
        objv = (int)edges[eb + 5];
        qrv  = q_rel[qrv];
    }

    const int l4 = lane * 4;  // this lane's 4-column slice of a 128-wide row
    // ---- gather hr / h_qr / hs, stage bf16 (+ hs f32) in LDS ----
    if (active) {
        for (int e = 0; e < 16; ++e) {
            int sub = __shfl(subv, e, 32);
            int rel = __shfl(relv, e, 32);
            int qr  = __shfl(qrv , e, 32);
            float4 vhs = *(const float4*)(hidden + (long long)sub * IN_DIM + l4);
            float4 vhr = *(const float4*)(rela   + (long long)rel * IN_DIM + l4);
            float4 vhq = *(const float4*)(rela   + (long long)qr  * IN_DIM + l4);
            uint2 hr2 = { pack2(vhr.x, vhr.y), pack2(vhr.z, vhr.w) };
            uint2 hq2 = { pack2(vhq.x, vhq.y), pack2(vhq.z, vhq.w) };
            uint2 hs2 = { pack2(vhs.x, vhs.y), pack2(vhs.z, vhs.w) };
            *(uint2*)(factors + e * 384 +       l4) = hr2;
            *(uint2*)(factors + e * 384 + 128 + l4) = hq2;
            *(uint2*)(factors + e * 384 + 256 + l4) = hs2;
            *(uint2*)(candin  + e * 256 +       l4) = hr2;
            *(float4*)(hsf + e * IN_DIM + l4) = vhs;
        }
    }
    __syncthreads();

    const int ncol = lane & 15;
    const int mh   = (lane >> 4) * 8;

    // ---- gates = sigmoid(factors @ gateW^T + b): 16 N-tiles x 12 K-steps ----
    if (active) {
        // A fragments are invariant across the N loop: cache in VGPRs (12 x 8 = 96 VGPRs)
        v16bf aF[12];
        #pragma unroll
        for (int ks = 0; ks < 12; ++ks) aF[ks] = load_frag(factors, 384, ks * 32, lane);
        for (int nt = 0; nt < 16; ++nt) {
            v8f acc = {};
            const unsigned short* Wb = gateWbf + nt * 16 * 384;
            #pragma unroll
            for (int ks = 0; ks < 12; ++ks) {
                v16bf b = load_frag(Wb, 384, ks * 32, lane);
                acc = WMMA_BF16(aF[ks], b, acc);
            }
            int n = nt * 16 + ncol;
            float bias = gateW_b[n];
            #pragma unroll
            for (int r = 0; r < 8; ++r)
                gatesL[(r + mh) * 256 + n] = sigm(acc[r] + bias);
        }
    }
    __syncthreads();

    // ---- candin second half = bf16(reset * hs) ----
    if (active) {
        for (int e = 0; e < 16; ++e) {
            float4 h = *(const float4*)(hsf + e * IN_DIM + l4);
            const float* rp = gatesL + e * 256 + 128 + l4;
            uint2 o = { pack2(rp[0] * h.x, rp[1] * h.y), pack2(rp[2] * h.z, rp[3] * h.w) };
            *(uint2*)(candin + e * 256 + 128 + l4) = o;
        }
    }
    __syncthreads();

    // ---- cand = tanh(candin @ ht^T + b); message = (1-u)*hs + u*cand ----
    if (active) {
        v16bf aC[8];
        #pragma unroll
        for (int ks = 0; ks < 8; ++ks) aC[ks] = load_frag(candin, 256, ks * 32, lane);
        for (int nt = 0; nt < 8; ++nt) {
            v8f acc = {};
            const unsigned short* Wb = htbf + nt * 16 * 256;
            #pragma unroll
            for (int ks = 0; ks < 8; ++ks) {
                v16bf b = load_frag(Wb, 256, ks * 32, lane);
                acc = WMMA_BF16(aC[ks], b, acc);
            }
            int n = nt * 16 + ncol;
            float bias = ht_b[n];
            #pragma unroll
            for (int r = 0; r < 8; ++r) {
                int m = r + mh;
                float cand = tanhf(acc[r] + bias);
                float u = gatesL[m * 256 + n];
                float h = hsf[m * IN_DIM + n];
                msgL[m * IN_DIM + n] = (1.0f - u) * h + u * cand;
            }
        }
    }
    __syncthreads();

    // ---- pre = hs@Ws^T + hr@Wr^T + h_qr@Wqr^T + b; leaky_relu ----
    if (active) {
        v16bf aS[4], aR[4], aQ[4];
        #pragma unroll
        for (int ks = 0; ks < 4; ++ks) {
            aS[ks] = load_frag(factors + 256, 384, ks * 32, lane);  // hs
            aR[ks] = load_frag(factors,       384, ks * 32, lane);  // hr
            aQ[ks] = load_frag(factors + 128, 384, ks * 32, lane);  // h_qr
        }
        for (int nt = 0; nt < 4; ++nt) {
            v8f acc = {};
            #pragma unroll
            for (int ks = 0; ks < 4; ++ks) {
                v16bf b = load_frag(Wsbf + nt * 16 * 128, 128, ks * 32, lane);
                acc = WMMA_BF16(aS[ks], b, acc);
                b = load_frag(Wrbf + nt * 16 * 128, 128, ks * 32, lane);
                acc = WMMA_BF16(aR[ks], b, acc);
                b = load_frag(Wqrbf + nt * 16 * 128, 128, ks * 32, lane);
                acc = WMMA_BF16(aQ[ks], b, acc);
            }
            int n = nt * 16 + ncol;
            float bias = Wqr_b[n];
            #pragma unroll
            for (int r = 0; r < 8; ++r) {
                float x = acc[r] + bias;
                preL[(r + mh) * ATTN_DIM + n] = (x > 0.0f) ? x : 0.01f * x;
            }
        }
    }
    __syncthreads();

    // ---- alpha = sigmoid(pre . walpha) per edge ----
    if (active && lane < 16) {
        float s = 0.0f;
        #pragma unroll 8
        for (int c = 0; c < ATTN_DIM; ++c) s += preL[lane * ATTN_DIM + c] * walpha[c];
        alphaL[lane] = sigm(s);
    }
    __syncthreads();

    // ---- scatter: agg[obj] += alpha * message ; deg[obj] += 1 ----
    if (active) {
        for (int e = 0; e < 16; ++e) {
            int obj = __shfl(objv, e, 32);
            float a = alphaL[e];
            float4 mv = *(const float4*)(msgL + e * IN_DIM + l4);
            float* dst = agg + (long long)obj * IN_DIM + l4;
            unsafeAtomicAdd(dst + 0, a * mv.x);
            unsafeAtomicAdd(dst + 1, a * mv.y);
            unsafeAtomicAdd(dst + 2, a * mv.z);
            unsafeAtomicAdd(dst + 3, a * mv.w);
            if (lane == 0) unsafeAtomicAdd(deg + obj, 1.0f);
        }
    }
}

// ---------------- output kernel: (agg / sqrt(deg+1e-4)) @ Wh^T ----------------
__global__ void __launch_bounds__(128) out_kernel(
    const unsigned short* __restrict__ Whbf, const float* __restrict__ agg,
    const float* __restrict__ deg, float* __restrict__ out)
{
    extern __shared__ char smem[];
    const int lane = threadIdx.x & 31;
    const int wave = threadIdx.x >> 5;
    const int tile = blockIdx.x * 4 + wave;
    const bool active = tile < NODE_TILES;
    unsigned short* A = (unsigned short*)(smem + (size_t)wave * (16 * IN_DIM * 2));
    const int l4 = lane * 4;

    if (active) {
        for (int e = 0; e < 16; ++e) {
            int node = tile * 16 + e;
            float sc = rsqrtf(deg[node] + 0.0001f);
            float4 v = *(const float4*)(agg + (long long)node * IN_DIM + l4);
            uint2 o = { pack2(v.x * sc, v.y * sc), pack2(v.z * sc, v.w * sc) };
            *(uint2*)(A + e * IN_DIM + l4) = o;
        }
    }
    __syncthreads();

    if (active) {
        const int ncol = lane & 15;
        const int mh   = (lane >> 4) * 8;
        v16bf aF[4];
        #pragma unroll
        for (int ks = 0; ks < 4; ++ks) aF[ks] = load_frag(A, 128, ks * 32, lane);
        for (int nt = 0; nt < 8; ++nt) {
            v8f acc = {};
            #pragma unroll
            for (int ks = 0; ks < 4; ++ks) {
                v16bf b = load_frag(Whbf + nt * 16 * 128, 128, ks * 32, lane);
                acc = WMMA_BF16(aF[ks], b, acc);
            }
            int n = nt * 16 + ncol;
            #pragma unroll
            for (int r = 0; r < 8; ++r)
                out[(long long)(tile * 16 + r + mh) * IN_DIM + n] = acc[r];
        }
    }
}

extern "C" void kernel_launch(void* const* d_in, const int* in_sizes, int n_in,
                              void* d_out, int out_size, void* d_ws, size_t ws_size,
                              hipStream_t stream) {
    (void)in_sizes; (void)n_in; (void)out_size; (void)ws_size;
    const int*       q_rel   = (const int*)d_in[1];
    const float*     hidden  = (const float*)d_in[2];
    const long long* edges   = (const long long*)d_in[3];
    const float*     rela    = (const float*)d_in[5];
    const float*     Ws      = (const float*)d_in[6];
    const float*     Wr      = (const float*)d_in[7];
    const float*     Wqr     = (const float*)d_in[8];
    const float*     Wqr_b   = (const float*)d_in[9];
    const float*     walpha  = (const float*)d_in[10];
    const float*     gateW   = (const float*)d_in[11];
    const float*     gateW_b = (const float*)d_in[12];
    const float*     htW     = (const float*)d_in[13];
    const float*     ht_b    = (const float*)d_in[14];
    const float*     Wh      = (const float*)d_in[15];

    char* ws = (char*)d_ws;
    float* agg = (float*)(ws + OFF_AGG);
    float* deg = (float*)(ws + OFF_DEG);
    unsigned short* gateWbf = (unsigned short*)(ws + OFF_GATEW);
    unsigned short* htbf    = (unsigned short*)(ws + OFF_HT);
    unsigned short* Wsbf    = (unsigned short*)(ws + OFF_WS);
    unsigned short* Wrbf    = (unsigned short*)(ws + OFF_WR);
    unsigned short* Wqrbf   = (unsigned short*)(ws + OFF_WQR);
    unsigned short* Whbf    = (unsigned short*)(ws + OFF_WH);

    // zero agg + deg (contiguous: 6,450,000 floats)
    zero_f32_kernel<<<2048, 256, 0, stream>>>(agg, 6450000);
    // weight conversions f32 -> bf16
    cvt_bf16_kernel<<<(98304 + 255) / 256, 256, 0, stream>>>(gateW, gateWbf, 98304);
    cvt_bf16_kernel<<<(32768 + 255) / 256, 256, 0, stream>>>(htW,   htbf,   32768);
    cvt_bf16_kernel<<<(8192  + 255) / 256, 256, 0, stream>>>(Ws,    Wsbf,   8192);
    cvt_bf16_kernel<<<(8192  + 255) / 256, 256, 0, stream>>>(Wr,    Wrbf,   8192);
    cvt_bf16_kernel<<<(8192  + 255) / 256, 256, 0, stream>>>(Wqr,   Wqrbf,  8192);
    cvt_bf16_kernel<<<(16384 + 255) / 256, 256, 0, stream>>>(Wh,    Whbf,   16384);

    // edge phase: 31250 tiles, 4 waves/block
    edge_kernel<<<(EDGE_TILES + 3) / 4, 128, 4 * WAVE_LDS, stream>>>(
        q_rel, hidden, edges, rela, Wqr_b, walpha, gateW_b, ht_b,
        gateWbf, htbf, Wsbf, Wrbf, Wqrbf, agg, deg);

    // output phase: 3125 tiles
    out_kernel<<<(NODE_TILES + 3) / 4, 128, 4 * 16 * IN_DIM * 2, stream>>>(
        Whbf, agg, deg, (float*)d_out);
}
